// MultiHeadAttention_9045201125405
// MI455X (gfx1250) — matrix operand order repaired and checked by reading, not compiled, not run on hardware
//
#include <hip/hip_runtime.h>
#include <hip/hip_bf16.h>

// ---------------------------------------------------------------------------
// Types for CDNA5 WMMA (wave32, 16x16x32 bf16 -> f32)
// ---------------------------------------------------------------------------
typedef __bf16 bf16;
typedef __attribute__((ext_vector_type(16))) bf16  v16bf;
typedef __attribute__((ext_vector_type(8)))  bf16  v8bf;
typedef __attribute__((ext_vector_type(8)))  float v8f;

#define WMMA_BF16(A, B, C) \
    __builtin_amdgcn_wmma_f32_16x16x32_bf16(false, (A), false, (B), (short)0, (C), false, false)

// A-fragment (16x32, 16-bit), this code's contiguous k order: lanes 0-15 hold row=lane, K 0..15;
// lanes 16-31 hold row=lane&15, K 16..31.
// Caller passes p already offset by (half ? 16 : 0); we read p[0..15].
__device__ __forceinline__ v16bf load_frag_a(const bf16* __restrict__ p) {
    v8bf lo = *(const v8bf*)(p);
    v8bf hi = *(const v8bf*)(p + 8);
    v16bf r;
#pragma unroll
    for (int i = 0; i < 8; ++i) { r[i] = lo[i]; r[i + 8] = hi[i]; }
    return r;
}

// B-fragment (32x16, 16-bit), k order as in A: lane = column (lane&15); lanes 0-15 hold K 0..15,
// lanes 16-31 hold K 16..31. Caller passes p already offset by (half ? 16 : 0);
// we read 16 contiguous bf16.
__device__ __forceinline__ v16bf load_frag_b(const bf16* __restrict__ p) {
    v8bf lo = *(const v8bf*)(p);
    v8bf hi = *(const v8bf*)(p + 8);
    v16bf r;
#pragma unroll
    for (int i = 0; i < 8; ++i) { r[i] = lo[i]; r[i + 8] = hi[i]; }
    return r;
}

// ---------------------------------------------------------------------------
// Problem constants
// ---------------------------------------------------------------------------
#define BATCH   2
#define SEQ     2048
#define DMODEL  2048
#define HEADS   16
#define DK      128
#define NQKV    (3 * DMODEL)         // 6144
#define MROWS   (BATCH * SEQ)        // 4096
#define INV_SQRT_DK 0.08838834764831845f

// ---------------------------------------------------------------------------
// Kernel 0: fp32 -> bf16 copy
// ---------------------------------------------------------------------------
__global__ void k_cvt_bf16(const float* __restrict__ in, bf16* __restrict__ out, int n) {
    int i = blockIdx.x * blockDim.x + threadIdx.x;
    if (i < n) out[i] = (bf16)in[i];
}

// ---------------------------------------------------------------------------
// Kernel 1: transpose fp32 [R][C] -> bf16 [C][R] (32x32 LDS tiles)
// ---------------------------------------------------------------------------
__global__ void k_transpose_bf16(const float* __restrict__ in, bf16* __restrict__ out,
                                 int R, int C) {
    __shared__ float tile[32][33];
    const int c0 = blockIdx.x * 32;
    const int r0 = blockIdx.y * 32;
    const int tx = threadIdx.x;      // 0..31
    const int ty = threadIdx.y;      // 0..7
#pragma unroll
    for (int i = 0; i < 32; i += 8)
        tile[ty + i][tx] = in[(size_t)(r0 + ty + i) * C + (c0 + tx)];
    __syncthreads();
#pragma unroll
    for (int i = 0; i < 32; i += 8)
        out[(size_t)(c0 + ty + i) * R + (r0 + tx)] = (bf16)tile[tx][ty + i];
}

// ---------------------------------------------------------------------------
// Shared GEMM tile core: one wave computes 16 (M) x 64 (N), K-contiguous bf16
// operands (A row-major [M][K], Wt row-major [N][K]).
// ---------------------------------------------------------------------------
__device__ __forceinline__ void gemm_wave_16x64(
    const bf16* __restrict__ A, const bf16* __restrict__ Wt,
    int K, int m0, int n0, int lane, v8f acc[4]) {
    const int half = lane >> 4;
    const int l15  = lane & 15;
    const bf16* pa = A + (size_t)(m0 + l15) * K + (half ? 16 : 0);
    const bf16* pb0 = Wt + (size_t)(n0 +  0 + l15) * K + (half ? 16 : 0);
    const bf16* pb1 = Wt + (size_t)(n0 + 16 + l15) * K + (half ? 16 : 0);
    const bf16* pb2 = Wt + (size_t)(n0 + 32 + l15) * K + (half ? 16 : 0);
    const bf16* pb3 = Wt + (size_t)(n0 + 48 + l15) * K + (half ? 16 : 0);
    for (int k = 0; k < K; k += 32) {
        v16bf a = load_frag_a(pa); pa += 32;
        v16bf b0 = load_frag_b(pb0); pb0 += 32;
        acc[0] = WMMA_BF16(a, b0, acc[0]);
        v16bf b1 = load_frag_b(pb1); pb1 += 32;
        acc[1] = WMMA_BF16(a, b1, acc[1]);
        v16bf b2 = load_frag_b(pb2); pb2 += 32;
        acc[2] = WMMA_BF16(a, b2, acc[2]);
        v16bf b3 = load_frag_b(pb3); pb3 += 32;
        acc[3] = WMMA_BF16(a, b3, acc[3]);
    }
}

// ---------------------------------------------------------------------------
// Kernel 2: QKV projection.  C[M=4096][N=6144] = Xb @ Wqkv_t^T + b.
// Column n decodes as head h = n/384, r = n%384:
//   r in [0,128)   -> Q  (scaled by 1/sqrt(dk)) -> Qs [B,H,S,DK] bf16
//   r in [128,256) -> K                         -> Kb [B,H,S,DK] bf16
//   r in [256,384) -> V  (transposed)           -> Vt [B,H,DK,S] bf16
// Block: 256 threads = 8 waves -> 128 x 64 tile.
// ---------------------------------------------------------------------------
__global__ __launch_bounds__(256) void k_gemm_qkv(
    const bf16* __restrict__ Xb, const bf16* __restrict__ Wt,
    const float* __restrict__ bias,
    bf16* __restrict__ Qs, bf16* __restrict__ Kb, bf16* __restrict__ Vt) {
    const int lane = threadIdx.x & 31;
    const int wave = threadIdx.x >> 5;
    const int m0 = blockIdx.x * 128 + wave * 16;
    const int n0 = blockIdx.y * 64;
    const int half = lane >> 4;
    const int l15  = lane & 15;

    v8f acc[4] = {};
    gemm_wave_16x64(Xb, Wt, DMODEL, m0, n0, lane, acc);

    const int mbase = m0 + (half ? 8 : 0);
    const int bidx  = mbase >> 11;       // batch
    const int sbase = mbase & (SEQ - 1); // sequence position
#pragma unroll
    for (int f = 0; f < 4; ++f) {
        const int n = n0 + f * 16 + l15;
        const int h = n / 384;
        const int r = n % 384;
        const float bb = bias[n];
        if (r < 128) {
            bf16* q = Qs + (((size_t)bidx * HEADS + h) * SEQ + sbase) * DK + r;
#pragma unroll
            for (int i = 0; i < 8; ++i)
                q[(size_t)i * DK] = (bf16)((acc[f][i] + bb) * INV_SQRT_DK);
        } else if (r < 256) {
            bf16* kp = Kb + (((size_t)bidx * HEADS + h) * SEQ + sbase) * DK + (r - 128);
#pragma unroll
            for (int i = 0; i < 8; ++i)
                kp[(size_t)i * DK] = (bf16)(acc[f][i] + bb);
        } else {
            bf16* vp = Vt + (((size_t)bidx * HEADS + h) * DK + (r - 256)) * (size_t)SEQ + sbase;
#pragma unroll
            for (int i = 0; i < 8; ++i)
                vp[i] = (bf16)(acc[f][i] + bb);
        }
    }
}

// ---------------------------------------------------------------------------
// Kernel 3: flash attention.  One wave handles a 16-row q-tile of one (b,h).
// Online softmax in f32; P bounced through per-wave LDS (f32 C-frag -> bf16
// A-frag re-layout); O accumulated in 8 f32 fragments (16 x 128).
// ---------------------------------------------------------------------------
#define AST 40  // LDS row stride in bf16 elements (80 B, keeps 16B alignment)

__global__ __launch_bounds__(128) void k_attn(
    const bf16* __restrict__ Qs, const bf16* __restrict__ Kb,
    const bf16* __restrict__ Vt, bf16* __restrict__ ctx) {
    __shared__ bf16 plds[4 * 16 * AST];

    const int lane  = threadIdx.x & 31;
    const int wslot = threadIdx.x >> 5;
    const int wid   = blockIdx.x * 4 + wslot;   // 0..4095
    const int bh    = wid >> 7;                 // 0..31
    const int qt    = wid & 127;
    const int half  = lane >> 4;
    const int l15   = lane & 15;
    const int q0    = qt * 16;

    const bf16* Qbh = Qs + (size_t)bh * SEQ * DK;
    const bf16* Kbh = Kb + (size_t)bh * SEQ * DK;
    const bf16* Vbh = Vt + (size_t)bh * DK * SEQ;
    bf16* myl = plds + wslot * 16 * AST;

    // Q fragments: 16 x 128 bf16 kept in registers (4 A-frags).
    v16bf qf[4];
#pragma unroll
    for (int t = 0; t < 4; ++t)
        qf[t] = load_frag_a(Qbh + (size_t)(q0 + l15) * DK + t * 32 + (half ? 16 : 0));

    v8f o[8] = {};
    float mi[8], li[8];
#pragma unroll
    for (int i = 0; i < 8; ++i) { mi[i] = -1e30f; li[i] = 0.0f; }

    for (int j0 = 0; j0 < SEQ; j0 += 32) {
        // ---- scores: two 16x16 tiles (columns j0..j0+15 and j0+16..j0+31)
        v8f s0 = {}, s1 = {};
#pragma unroll
        for (int t = 0; t < 4; ++t) {
            v16bf kf0 = load_frag_b(Kbh + (size_t)(j0      + l15) * DK + t * 32 + (half ? 16 : 0));
            s0 = WMMA_BF16(qf[t], kf0, s0);
            v16bf kf1 = load_frag_b(Kbh + (size_t)(j0 + 16 + l15) * DK + t * 32 + (half ? 16 : 0));
            s1 = WMMA_BF16(qf[t], kf1, s1);
        }

        // ---- online softmax (rows split: lanes<16 rows 0-7, lanes>=16 rows 8-15)
        float al[8];
#pragma unroll
        for (int i = 0; i < 8; ++i) {
            float v = fmaxf(s0[i], s1[i]);
            v = fmaxf(v, __shfl_xor(v, 1, 32));
            v = fmaxf(v, __shfl_xor(v, 2, 32));
            v = fmaxf(v, __shfl_xor(v, 4, 32));
            v = fmaxf(v, __shfl_xor(v, 8, 32));
            const float mn = fmaxf(mi[i], v);
            al[i] = __expf(mi[i] - mn);
            mi[i] = mn;
            const float p0 = __expf(s0[i] - mn);
            const float p1 = __expf(s1[i] - mn);
            s0[i] = p0; s1[i] = p1;
            float rs = p0 + p1;
            rs += __shfl_xor(rs, 1, 32);
            rs += __shfl_xor(rs, 2, 32);
            rs += __shfl_xor(rs, 4, 32);
            rs += __shfl_xor(rs, 8, 32);
            li[i] = li[i] * al[i] + rs;
        }

        // ---- rescale O accumulators
#pragma unroll
        for (int f = 0; f < 8; ++f)
#pragma unroll
            for (int i = 0; i < 8; ++i) o[f][i] *= al[i];

        // ---- write P (bf16) to per-wave LDS in row-major 16x32
        {
            const int rb = half ? 8 : 0;
#pragma unroll
            for (int i = 0; i < 8; ++i) {
                myl[(rb + i) * AST + l15]      = (bf16)s0[i];
                myl[(rb + i) * AST + l15 + 16] = (bf16)s1[i];
            }
        }
        // ---- reread as A-fragment (same-wave DS ops are in-order)
        v16bf pf;
        {
            const bf16* pp = myl + l15 * AST + (half ? 16 : 0);
            v8bf lo = *(const v8bf*)(pp);
            v8bf hi = *(const v8bf*)(pp + 8);
#pragma unroll
            for (int i = 0; i < 8; ++i) { pf[i] = lo[i]; pf[i + 8] = hi[i]; }
        }

        // ---- O += P @ V  (V pre-transposed: Vt[d][s], contiguous along s)
#pragma unroll
        for (int f = 0; f < 8; ++f) {
            v16bf vf = load_frag_b(Vbh + (size_t)(f * 16 + l15) * SEQ + j0 + (half ? 16 : 0));
            o[f] = WMMA_BF16(pf, vf, o[f]);
        }
    }

    // ---- epilogue: ctx[b][s][h*128 + d] bf16 (heads re-merged)
    const int b = bh >> 4;
    const int h = bh & 15;
    const int sbase = q0 + (half ? 8 : 0);
#pragma unroll
    for (int f = 0; f < 8; ++f) {
#pragma unroll
        for (int i = 0; i < 8; ++i) {
            const float val = o[f][i] / li[i];
            ctx[((size_t)b * SEQ + sbase + i) * DMODEL + h * DK + f * 16 + l15] = (bf16)val;
        }
    }
}

// ---------------------------------------------------------------------------
// Kernel 4: output projection.  out[M=4096][2048] f32 = ctx @ Wout_t^T + b.
// ---------------------------------------------------------------------------
__global__ __launch_bounds__(256) void k_gemm_out(
    const bf16* __restrict__ A, const bf16* __restrict__ Wt,
    const float* __restrict__ bias, float* __restrict__ out) {
    const int lane = threadIdx.x & 31;
    const int wave = threadIdx.x >> 5;
    const int m0 = blockIdx.x * 128 + wave * 16;
    const int n0 = blockIdx.y * 64;
    const int half = lane >> 4;
    const int l15  = lane & 15;

    v8f acc[4] = {};
    gemm_wave_16x64(A, Wt, DMODEL, m0, n0, lane, acc);

    const int mbase = m0 + (half ? 8 : 0);
#pragma unroll
    for (int f = 0; f < 4; ++f) {
        const int n = n0 + f * 16 + l15;
        const float bb = bias[n];
        float* po = out + (size_t)mbase * DMODEL + n;
#pragma unroll
        for (int i = 0; i < 8; ++i)
            po[(size_t)i * DMODEL] = acc[f][i] + bb;
    }
}

// ---------------------------------------------------------------------------
// Host launcher
// ---------------------------------------------------------------------------
extern "C" void kernel_launch(void* const* d_in, const int* in_sizes, int n_in,
                              void* d_out, int out_size, void* d_ws, size_t ws_size,
                              hipStream_t stream) {
    (void)in_sizes; (void)n_in; (void)out_size; (void)ws_size;

    const float* x     = (const float*)d_in[0];  // [B,S,D]
    const float* Wqkv  = (const float*)d_in[1];  // [D, 3D]
    const float* bqkv  = (const float*)d_in[2];  // [3D]
    const float* Wout  = (const float*)d_in[3];  // [D, D]
    const float* bout  = (const float*)d_in[4];  // [D]
    float* out = (float*)d_out;                  // [B,S,D] f32

    char* ws = (char*)d_ws;
    // byte offsets (all 256B-aligned)
    bf16* Xb    = (bf16*)(ws);                         // 4096*2048       -> 16 MB
    bf16* Wqkvt = (bf16*)(ws + (16u  << 20));          // 6144*2048       -> 24 MB
    bf16* Woutt = (bf16*)(ws + (40u  << 20));          // 2048*2048       ->  8 MB
    bf16* Qs    = (bf16*)(ws + (48u  << 20));          // 32*2048*128     -> 16 MB
    bf16* Kb    = (bf16*)(ws + (64u  << 20));          // 32*2048*128     -> 16 MB
    bf16* Vt    = (bf16*)(ws + (80u  << 20));          // 32*128*2048     -> 16 MB
    bf16* ctx   = Xb;  // Xb is dead after k_gemm_qkv; reuse for context

    // 0) x -> bf16
    {
        const int n = MROWS * DMODEL;
        k_cvt_bf16<<<(n + 255) / 256, 256, 0, stream>>>(x, Xb, n);
    }
    // 1) weight transposes (fp32 [K][N] -> bf16 [N][K])
    {
        dim3 tb(32, 8);
        k_transpose_bf16<<<dim3(NQKV / 32, DMODEL / 32), tb, 0, stream>>>(Wqkv, Wqkvt, DMODEL, NQKV);
        k_transpose_bf16<<<dim3(DMODEL / 32, DMODEL / 32), tb, 0, stream>>>(Wout, Woutt, DMODEL, DMODEL);
    }
    // 2) QKV projection + head split (+ V transpose, Q pre-scale)
    k_gemm_qkv<<<dim3(MROWS / 128, NQKV / 64), 256, 0, stream>>>(Xb, Wqkvt, bqkv, Qs, Kb, Vt);
    // 3) flash attention (4096 q-tiles of 16 rows; 4 waves / block)
    k_attn<<<(BATCH * HEADS * (SEQ / 16)) / 4, 128, 0, stream>>>(Qs, Kb, Vt, ctx);
    // 4) output projection -> f32
    k_gemm_out<<<dim3(MROWS / 128, DMODEL / 64), 256, 0, stream>>>(ctx, Woutt, bout, out);
}
